// EncoderLayer_21457656611105
// MI455X (gfx1250) — matrix operand order repaired and checked by reading, not compiled, not run on hardware
//
#include <hip/hip_runtime.h>
#include <hip/hip_bf16.h>

// ---------------- problem constants ----------------
constexpr int  cB    = 8;
constexpr int  cT    = 256;
constexpr int  cM    = 64;
constexpr int  cF    = 128;
constexpr int  cH    = 8;
constexpr int  cDFF  = 512;
constexpr int  cNTOK = cB * cT * cM;                 // 131072 tokens
constexpr long long cNX = (long long)cNTOK * cF;     // 16,777,216 elems

// ---------------- types for WMMA ----------------
typedef __bf16 bf16_t;
typedef __attribute__((ext_vector_type(16))) __bf16 v16bf;
typedef __attribute__((ext_vector_type(8)))  __bf16 v8bf;
typedef __attribute__((ext_vector_type(8)))  float  v8f;
typedef int v4i_g __attribute__((vector_size(16)));  // matches builtin param type

union FragAB { v16bf v; v8bf h[2]; uint4 q[2]; };

// ---------------- CDNA5 async LDS staging (probe-guarded) ----------------
#if defined(__gfx1250__) && __has_builtin(__builtin_amdgcn_global_load_async_to_lds_b128)
#define USE_ASYNC_LDS 1
#else
#define USE_ASYNC_LDS 0
#endif

__device__ __forceinline__ void stage_b128(const bf16_t* g, void* l) {
#if USE_ASYNC_LDS
  __builtin_amdgcn_global_load_async_to_lds_b128(
      (__attribute__((address_space(1))) v4i_g*)(void*)g,
      (__attribute__((address_space(3))) v4i_g*)l,
      0, 0);
#else
  *(uint4*)l = *(const uint4*)g;   // synchronous fallback
#endif
}

__device__ __forceinline__ void wait_async_all() {
#if USE_ASYNC_LDS
#if __has_builtin(__builtin_amdgcn_s_wait_asynccnt)
  __builtin_amdgcn_s_wait_asynccnt(0);
#else
  asm volatile("s_wait_asynccnt 0x0" ::: "memory");
#endif
#endif
}

// LDS B-tile: 64 cols x 32 K bf16; row stride 80B -> conflict-free frag reads
constexpr int LDSB_STRIDE = 80;
constexpr int LDSB_BYTES  = 64 * LDSB_STRIDE;        // 5120 B per buffer

// A 16x32 bf16 fragment: lane(kh) holds K = kh*16+[0..15]
__device__ __forceinline__ void load_a_frag(FragAB& a, const bf16_t* aptr, int kb, int kh) {
  a.h[0] = *(const v8bf*)(aptr + kb + kh * 16);
  a.h[1] = *(const v8bf*)(aptr + kb + 8 + kh * 16);
}

// 4 x (B-frag read from LDS + WMMA).  brd = &buf[(lane&15)*80 + kh*32]
__device__ __forceinline__ void compute_tile(v8f* acc, const FragAB& a, const unsigned char* brd) {
#pragma unroll
  for (int s = 0; s < 4; ++s) {
    FragAB b;
    b.h[0] = *(const v8bf*)(brd + s * 16 * LDSB_STRIDE);
    b.h[1] = *(const v8bf*)(brd + s * 16 * LDSB_STRIDE + 16);
    acc[s] = __builtin_amdgcn_wmma_f32_16x16x32_bf16(
        false, a.v, false, b.v, (short)0, acc[s], false, false);
  }
}

// =====================================================================
// Generic batched bf16 WMMA GEMM:  C[z] = act( scale * A[z] * Bt[z]^T + bias )
// Requirements (guaranteed by all launches): Ncols%64==0, Mrows%16==0, K%64==0.
// Block: 256 threads = 8 waves; wave w -> rows 16, block cols 64.
// B tiles double-buffered in LDS via async-to-LDS; A frags double-buffered in regs.
// =====================================================================
__global__ __launch_bounds__(256) void gemm_bf16_wmma(
    const bf16_t* __restrict__ A, const bf16_t* __restrict__ Bt,
    const float* __restrict__ bias,
    float* __restrict__ Cf, bf16_t* __restrict__ Cb,
    int Mrows, int Ncols, int K,
    int lda, int ldb, int ldc,
    int bdiv,
    long long sA1, long long sA2, long long sB1, long long sB2,
    long long sC1, long long sC2,
    float scale, int relu)
{
  __shared__ __align__(16) unsigned char ldsB[2][LDSB_BYTES];

  const int tid  = threadIdx.x;
  const int lane = tid & 31;
  const int wave = tid >> 5;
  const int z  = blockIdx.z;
  const int z1 = z / bdiv;
  const int z2 = z - z1 * bdiv;
  A  += z1 * sA1 + z2 * sA2;
  Bt += z1 * sB1 + z2 * sB2;
  const long long coff = z1 * sC1 + z2 * sC2;

  const int rowBase = blockIdx.x * 128 + wave * 16;
  const int colBase = blockIdx.y * 64;
  const int r   = lane & 15;
  const int kh  = lane >> 4;
  const bool aok = (rowBase + r) < Mrows;            // wave-uniform (Mrows%16==0)
  const int arow = aok ? (rowBase + r) : (Mrows - 1);  // branchless clamp
  const bf16_t* aptr = A + (long long)arow * lda;

  // B staging: each of 256 threads copies one 16B chunk per K-tile
  const int trow = tid >> 2;                          // 0..63
  const int tseg = tid & 3;                           // 0..3
  const bf16_t* bsrc = Bt + (long long)(colBase + trow) * ldb + tseg * 8;
  unsigned char* bw0 = &ldsB[0][trow * LDSB_STRIDE + tseg * 16];
  unsigned char* bw1 = &ldsB[1][trow * LDSB_STRIDE + tseg * 16];
  const unsigned char* br0 = &ldsB[0][(lane & 15) * LDSB_STRIDE + kh * 32];
  const unsigned char* br1 = &ldsB[1][(lane & 15) * LDSB_STRIDE + kh * 32];

  v8f acc[4];
#pragma unroll
  for (int s = 0; s < 4; ++s)
#pragma unroll
    for (int i = 0; i < 8; ++i) acc[s][i] = 0.0f;

  const int nk = K >> 5;                              // even

  FragAB a0, a1;
  stage_b128(bsrc, bw0);
  load_a_frag(a0, aptr, 0, kh);
  wait_async_all();
  __syncthreads();

  for (int i = 0; i < nk; i += 2) {
    stage_b128(bsrc + (i + 1) * 32, bw1);
    load_a_frag(a1, aptr, (i + 1) * 32, kh);
    if (((i & 7) == 0) && ((i + 8) * 32 < K)) {
      __builtin_prefetch(aptr + (i + 8) * 32, 0, 1);
      __builtin_prefetch(bsrc + (i + 8) * 32, 0, 1);
    }
    compute_tile(acc, a0, br0);
    wait_async_all();
    __syncthreads();
    if (i + 2 < nk) {
      stage_b128(bsrc + (i + 2) * 32, bw0);
      load_a_frag(a0, aptr, (i + 2) * 32, kh);
    }
    compute_tile(acc, a1, br1);
    wait_async_all();
    __syncthreads();
  }

  if (!aok) return;   // after all barriers; wave-uniform
  const int col0  = colBase + (lane & 15);
  const int rbase = rowBase + (kh << 3);
  if (Cf) {
#pragma unroll
    for (int s = 0; s < 4; ++s) {
      const int col = col0 + s * 16;
      const float bv = bias ? bias[col] : 0.0f;
#pragma unroll
      for (int i = 0; i < 8; ++i) {
        float vv = acc[s][i] * scale + bv;
        if (relu) vv = fmaxf(vv, 0.0f);
        Cf[coff + (long long)(rbase + i) * ldc + col] = vv;
      }
    }
  } else {
#pragma unroll
    for (int s = 0; s < 4; ++s) {
      const int col = col0 + s * 16;
      const float bv = bias ? bias[col] : 0.0f;
#pragma unroll
      for (int i = 0; i < 8; ++i) {
        float vv = acc[s][i] * scale + bv;
        if (relu) vv = fmaxf(vv, 0.0f);
        Cb[coff + (long long)(rbase + i) * ldc + col] = (bf16_t)vv;
      }
    }
  }
}

// =====================================================================
// 3x3 SAME conv over (T,M), F(128)->128, expressed as one K=1152 GEMM:
// Bt = wT9 [128 out, 9*128 in]; A rows are 9 shifted x-rows with branchless
// zero-masking for padding (uint AND, no exec divergence).
// =====================================================================
__global__ __launch_bounds__(256) void conv3x3_wmma(
    const bf16_t* __restrict__ xbf, const bf16_t* __restrict__ wT9,
    const float* __restrict__ bias, float* __restrict__ out)
{
  __shared__ __align__(16) unsigned char ldsB[2][LDSB_BYTES];

  const int tid  = threadIdx.x;
  const int lane = tid & 31;
  const int wave = tid >> 5;
  const int rowBase = blockIdx.x * 128 + wave * 16;
  const int colBase = blockIdx.y * 64;
  const int r  = lane & 15;
  const int kh = lane >> 4;
  const int row = rowBase + r;                        // always < NTOK
  const int b = row >> 14;
  const int t = (row >> 6) & 255;
  const int m = row & 63;

  const int trow = tid >> 2;
  const int tseg = tid & 3;
  const bf16_t* bsrc = wT9 + (long long)(colBase + trow) * 1152 + tseg * 8;
  unsigned char* bw0 = &ldsB[0][trow * LDSB_STRIDE + tseg * 16];
  unsigned char* bw1 = &ldsB[1][trow * LDSB_STRIDE + tseg * 16];
  const unsigned char* br0 = &ldsB[0][(lane & 15) * LDSB_STRIDE + kh * 32];
  const unsigned char* br1 = &ldsB[1][(lane & 15) * LDSB_STRIDE + kh * 32];

  v8f acc[4];
#pragma unroll
  for (int s = 0; s < 4; ++s)
#pragma unroll
    for (int i = 0; i < 8; ++i) acc[s][i] = 0.0f;

  // masked A loader: kglob in [0,1152) -> tap + in-channel
  auto load_a_conv = [&](FragAB& a, int kglob) {
    const int tap = kglob >> 7;
    const int kk  = kglob & 127;
    const int dt = tap / 3 - 1, dm = tap - (tap / 3) * 3 - 1;
    const int t2 = t + dt, m2 = m + dm;
    const bool ok = ((unsigned)t2 < 256u) && ((unsigned)m2 < 64u);
    const unsigned msk = ok ? 0xFFFFFFFFu : 0u;
    const int t2c = ok ? t2 : t;
    const int m2c = ok ? m2 : m;
    const bf16_t* ap = xbf + ((((long long)b << 8) + t2c) * 64 + m2c) * 128 + kk + kh * 16;
    FragAB u;
    u.h[0] = *(const v8bf*)(ap);
    u.h[1] = *(const v8bf*)(ap + 8);
#pragma unroll
    for (int w = 0; w < 2; ++w) {
      u.q[w].x &= msk; u.q[w].y &= msk; u.q[w].z &= msk; u.q[w].w &= msk;
    }
    a = u;
  };

  const int nk = 36;                                  // 1152/32, even

  FragAB a0, a1;
  stage_b128(bsrc, bw0);
  load_a_conv(a0, 0);
  wait_async_all();
  __syncthreads();

  for (int i = 0; i < nk; i += 2) {
    stage_b128(bsrc + (i + 1) * 32, bw1);
    load_a_conv(a1, (i + 1) * 32);
    compute_tile(acc, a0, br0);
    wait_async_all();
    __syncthreads();
    if (i + 2 < nk) {
      stage_b128(bsrc + (i + 2) * 32, bw0);
      load_a_conv(a0, (i + 2) * 32);
    }
    compute_tile(acc, a1, br1);
    wait_async_all();
    __syncthreads();
  }

  const int col0  = colBase + (lane & 15);
  const int rbase = rowBase + (kh << 3);
#pragma unroll
  for (int s = 0; s < 4; ++s) {
    const int col = col0 + s * 16;
    const float bv = bias[col];
#pragma unroll
    for (int i = 0; i < 8; ++i)
      out[(long long)(rbase + i) * 128 + col] = acc[s][i] + bv;
  }
}

// ----------------- wave-per-row softmax (f32 in -> bf16 out) -----------------
template <int L>
__global__ __launch_bounds__(256) void softmax_rows(
    const float* __restrict__ S, bf16_t* __restrict__ P, int rows)
{
  constexpr int PER = L / 32;
  const int gw = (int)(((long long)blockIdx.x * blockDim.x + threadIdx.x) >> 5);
  const int lane = threadIdx.x & 31;
  if (gw >= rows) return;
  const float* p = S + (long long)gw * L;
  float vv[PER];
  float mx = -3.0e38f;
#pragma unroll
  for (int i = 0; i < PER; ++i) { vv[i] = p[lane + i * 32]; mx = fmaxf(mx, vv[i]); }
#pragma unroll
  for (int d = 16; d >= 1; d >>= 1) mx = fmaxf(mx, __shfl_xor(mx, d, 32));
  float sum = 0.0f;
#pragma unroll
  for (int i = 0; i < PER; ++i) { vv[i] = __expf(vv[i] - mx); sum += vv[i]; }
#pragma unroll
  for (int d = 16; d >= 1; d >>= 1) sum += __shfl_xor(sum, d, 32);
  const float inv = 1.0f / sum;
  bf16_t* q = P + (long long)gw * L;
#pragma unroll
  for (int i = 0; i < PER; ++i) q[lane + i * 32] = (bf16_t)(vv[i] * inv);
}

// ----------------- residual add + LayerNorm (row = 128) -----------------
__global__ __launch_bounds__(256) void add_layernorm(
    const float* __restrict__ X, const float* __restrict__ Y,
    const float* __restrict__ g, const float* __restrict__ bt,
    float* __restrict__ outF, bf16_t* __restrict__ outB, int rows)
{
  const int gw = (int)(((long long)blockIdx.x * blockDim.x + threadIdx.x) >> 5);
  const int lane = threadIdx.x & 31;
  if (gw >= rows) return;
  const long long base = (long long)gw * 128 + lane * 4;
  const float4 x4 = *(const float4*)(X + base);
  const float4 y4 = *(const float4*)(Y + base);
  float s[4] = { x4.x + y4.x, x4.y + y4.y, x4.z + y4.z, x4.w + y4.w };
  float sum = s[0] + s[1] + s[2] + s[3];
  float sq  = s[0]*s[0] + s[1]*s[1] + s[2]*s[2] + s[3]*s[3];
#pragma unroll
  for (int d = 16; d >= 1; d >>= 1) {
    sum += __shfl_xor(sum, d, 32);
    sq  += __shfl_xor(sq,  d, 32);
  }
  const float mean = sum * (1.0f / 128.0f);
  const float var  = sq * (1.0f / 128.0f) - mean * mean;
  const float inv  = rsqrtf(var + 1e-6f);
#pragma unroll
  for (int j = 0; j < 4; ++j) {
    const int c = lane * 4 + j;
    const float o = (s[j] - mean) * inv * g[c] + bt[c];
    if (outF) outF[base + j] = o;
    if (outB) outB[base + j] = (bf16_t)o;
  }
}

// ----------------- elementwise convert / pack kernels -----------------
__global__ void convert_f32_bf16(const float* __restrict__ in, bf16_t* __restrict__ out, long long n) {
  long long i = (long long)blockIdx.x * blockDim.x + threadIdx.x;
  const long long stride = (long long)gridDim.x * blockDim.x;
  for (; i < n; i += stride) out[i] = (bf16_t)in[i];
}

// w[K,N] f32 -> wT[N,K] bf16
__global__ void transpose_wT_kernel(const float* __restrict__ w, bf16_t* __restrict__ wT,
                                    int K, int N) {
  const long long total = (long long)K * N;
  long long i = (long long)blockIdx.x * blockDim.x + threadIdx.x;
  const long long stride = (long long)gridDim.x * blockDim.x;
  for (; i < total; i += stride) {
    const int n = (int)(i / K);
    const int k = (int)(i - (long long)n * K);
    wT[i] = (bf16_t)w[(long long)k * N + n];
  }
}

// x[b,t,m,f] f32 -> xm[b,m,t,f] bf16
__global__ void pack_xm_kernel(const float* __restrict__ x, bf16_t* __restrict__ xm) {
  long long i = (long long)blockIdx.x * blockDim.x + threadIdx.x;
  const long long stride = (long long)gridDim.x * blockDim.x;
  for (; i < cNX; i += stride) {
    const int f = (int)(i & 127);
    const long long rr = i >> 7;
    const int t = (int)(rr & 255);
    const long long bm = rr >> 8;
    const int m = (int)(bm & 63);
    const int b = (int)(bm >> 6);
    xm[i] = (bf16_t)x[((((long long)b * 256 + t) * 64 + m) << 7) + f];
  }
}

// v[b,k,m,h,d] f32 -> vt[b,h,(m*16+d),k] bf16
__global__ void pack_vt_kernel(const float* __restrict__ v, bf16_t* __restrict__ vt) {
  const long long total = (long long)cB * cH * 1024 * 256;
  long long i = (long long)blockIdx.x * blockDim.x + threadIdx.x;
  const long long stride = (long long)gridDim.x * blockDim.x;
  for (; i < total; i += stride) {
    const int k = (int)(i & 255);
    const int c = (int)((i >> 8) & 1023);
    const int h = (int)((i >> 18) & 7);
    const int b = (int)(i >> 21);
    const int d = c & 15, m = c >> 4;
    vt[i] = (bf16_t)v[((((long long)b * 256 + k) * 64 + m) << 7) + h * 16 + d];
  }
}

// tmp[b,h,q,(n*16+d)] f32 -> tmp2[b,h,(q*16+d),n] bf16
__global__ void pack_tmp2_kernel(const float* __restrict__ tmp, bf16_t* __restrict__ tmp2) {
  const long long total = (long long)cB * cH * 4096 * 64;
  long long i = (long long)blockIdx.x * blockDim.x + threadIdx.x;
  const long long stride = (long long)gridDim.x * blockDim.x;
  for (; i < total; i += stride) {
    const int n  = (int)(i & 63);
    const int qd = (int)((i >> 6) & 4095);
    const int h  = (int)((i >> 18) & 7);
    const int b  = (int)(i >> 21);
    const int d = qd & 15, q = qd >> 4;
    tmp2[i] = (bf16_t)tmp[(((long long)(b * 8 + h) * 256 + q) << 10) + n * 16 + d];
  }
}

// attn2[b,h,(q*16+d),m] f32 -> rows[(b,q,m), h*16+d] bf16
__global__ void pack_attn_rows_kernel(const float* __restrict__ a2, bf16_t* __restrict__ ar) {
  long long i = (long long)blockIdx.x * blockDim.x + threadIdx.x;
  const long long stride = (long long)gridDim.x * blockDim.x;
  for (; i < cNX; i += stride) {
    const int c = (int)(i & 127);
    const int h = c >> 4, d = c & 15;
    const long long row = i >> 7;
    const int m = (int)(row & 63);
    const int q = (int)((row >> 6) & 255);
    const int b = (int)(row >> 14);
    ar[i] = (bf16_t)a2[(((long long)(b * 8 + h) * 4096 + q * 16 + d) << 6) + m];
  }
}

// =====================================================================
// Host-side orchestration
// =====================================================================
extern "C" void kernel_launch(void* const* d_in, const int* in_sizes, int n_in,
                              void* d_out, int out_size, void* d_ws, size_t ws_size,
                              hipStream_t stream) {
  (void)in_sizes; (void)n_in; (void)out_size; (void)ws_size;
  const float* x     = (const float*)d_in[0];
  const float* qt_w  = (const float*)d_in[1];
  const float* qt_b  = (const float*)d_in[2];
  const float* kt_w  = (const float*)d_in[3];
  const float* kt_b  = (const float*)d_in[4];
  const float* qm_w  = (const float*)d_in[5];
  const float* qm_b  = (const float*)d_in[6];
  const float* km_w  = (const float*)d_in[7];
  const float* km_b  = (const float*)d_in[8];
  const float* v_k   = (const float*)d_in[9];
  const float* v_b   = (const float*)d_in[10];
  const float* o_w   = (const float*)d_in[11];
  const float* o_b   = (const float*)d_in[12];
  const float* f1_w  = (const float*)d_in[13];
  const float* f1_b  = (const float*)d_in[14];
  const float* f2_w  = (const float*)d_in[15];
  const float* f2_b  = (const float*)d_in[16];
  const float* ln1_g = (const float*)d_in[17];
  const float* ln1_b = (const float*)d_in[18];
  const float* ln2_g = (const float*)d_in[19];
  const float* ln2_b = (const float*)d_in[20];
  float* outp = (float*)d_out;

  char* ws = (char*)d_ws;
  size_t off = 0;
  auto alloc = [&](size_t bytes) -> void* {
    void* p = ws + off;
    off += (bytes + 255) & ~(size_t)255;
    return p;
  };

  bf16_t* x_bf    = (bf16_t*)alloc(cNX * 2);
  bf16_t* xm_bf   = (bf16_t*)alloc(cNX * 2);
  bf16_t* qt_wT   = (bf16_t*)alloc((size_t)512 * 8192 * 2);
  bf16_t* kt_wT   = (bf16_t*)alloc((size_t)512 * 8192 * 2);
  bf16_t* qm_wT   = (bf16_t*)alloc((size_t)512 * 32768 * 2);
  bf16_t* km_wT   = (bf16_t*)alloc((size_t)512 * 32768 * 2);
  bf16_t* o_wT    = (bf16_t*)alloc((size_t)128 * 128 * 2);
  bf16_t* f1_wT   = (bf16_t*)alloc((size_t)512 * 128 * 2);
  bf16_t* f2_wT   = (bf16_t*)alloc((size_t)128 * 512 * 2);
  bf16_t* vkT     = (bf16_t*)alloc((size_t)128 * 1152 * 2);   // [out=128][9*128]
  bf16_t* qt_bf   = (bf16_t*)alloc((size_t)2048 * 512 * 2);
  bf16_t* kt_bf   = (bf16_t*)alloc((size_t)2048 * 512 * 2);
  bf16_t* qm_bf   = (bf16_t*)alloc((size_t)512 * 512 * 2);
  bf16_t* km_bf   = (bf16_t*)alloc((size_t)512 * 512 * 2);
  float*  v_f32   = (float*)alloc(cNX * 4);
  bf16_t* vt_bf   = (bf16_t*)alloc((size_t)cB * cH * 1024 * 256 * 2);
  float*  S_t     = (float*)alloc((size_t)cB * cH * 256 * 256 * 4);
  bf16_t* at_bf   = (bf16_t*)alloc((size_t)cB * cH * 256 * 256 * 2);
  float*  S_m     = (float*)alloc((size_t)cB * cH * 64 * 64 * 4);
  bf16_t* am_bf   = (bf16_t*)alloc((size_t)cB * cH * 64 * 64 * 2);
  float*  tmp_f   = (float*)alloc((size_t)cB * cH * 256 * 1024 * 4);
  bf16_t* tmp2_bf = (bf16_t*)alloc((size_t)cB * cH * 4096 * 64 * 2);
  float*  attn2_f = (float*)alloc((size_t)cB * cH * 4096 * 64 * 4);
  bf16_t* arow_bf = (bf16_t*)alloc(cNX * 2);
  float*  attn_o  = (float*)alloc(cNX * 4);
  float*  out1_f  = (float*)alloc(cNX * 4);
  bf16_t* out1_bf = (bf16_t*)alloc(cNX * 2);
  bf16_t* h1_bf   = (bf16_t*)alloc((size_t)cNTOK * cDFF * 2);
  float*  ffn_f   = (float*)alloc(cNX * 4);

  auto gemm = [&](const bf16_t* A, const bf16_t* Bt, const float* bias,
                  float* Cf, bf16_t* Cb, int Mr, int Nc, int K,
                  int lda, int ldb, int ldc, int nz, int bdiv,
                  long long sA1, long long sA2, long long sB1, long long sB2,
                  long long sC1, long long sC2, float scale, int relu) {
    dim3 grid((Mr + 127) / 128, Nc / 64, nz);
    gemm_bf16_wmma<<<grid, 256, 0, stream>>>(A, Bt, bias, Cf, Cb, Mr, Nc, K,
                                             lda, ldb, ldc, bdiv,
                                             sA1, sA2, sB1, sB2, sC1, sC2,
                                             scale, relu);
  };

  // ---- stage 0: precision conversion + weight transposes ----
  convert_f32_bf16<<<8192, 256, 0, stream>>>(x, x_bf, cNX);
  pack_xm_kernel<<<8192, 256, 0, stream>>>(x, xm_bf);
  transpose_wT_kernel<<<8192, 256, 0, stream>>>(qt_w, qt_wT, 8192, 512);
  transpose_wT_kernel<<<8192, 256, 0, stream>>>(kt_w, kt_wT, 8192, 512);
  transpose_wT_kernel<<<8192, 256, 0, stream>>>(qm_w, qm_wT, 32768, 512);
  transpose_wT_kernel<<<8192, 256, 0, stream>>>(km_w, km_wT, 32768, 512);
  transpose_wT_kernel<<<64,   256, 0, stream>>>(o_w,  o_wT,  128, 128);
  transpose_wT_kernel<<<256,  256, 0, stream>>>(f1_w, f1_wT, 128, 512);
  transpose_wT_kernel<<<256,  256, 0, stream>>>(f2_w, f2_wT, 512, 128);
  transpose_wT_kernel<<<576,  256, 0, stream>>>(v_k,  vkT,   1152, 128);  // conv as [128,1152]

  // ---- stage 1: Q/K projections (bf16 out, bias in f32) ----
  gemm(x_bf,  qt_wT, qt_b, nullptr, qt_bf, 2048, 512, 8192,  8192,  8192,  512, 1, 1, 0,0,0,0,0,0, 1.0f, 0);
  gemm(x_bf,  kt_wT, kt_b, nullptr, kt_bf, 2048, 512, 8192,  8192,  8192,  512, 1, 1, 0,0,0,0,0,0, 1.0f, 0);
  gemm(xm_bf, qm_wT, qm_b, nullptr, qm_bf, 512,  512, 32768, 32768, 32768, 512, 1, 1, 0,0,0,0,0,0, 1.0f, 0);
  gemm(xm_bf, km_wT, km_b, nullptr, km_bf, 512,  512, 32768, 32768, 32768, 512, 1, 1, 0,0,0,0,0,0, 1.0f, 0);

  // ---- stage 2: 3x3 value conv (single K=1152 WMMA GEMM) ----
  conv3x3_wmma<<<dim3(1024, 2, 1), 256, 0, stream>>>(x_bf, vkT, v_b, v_f32);
  pack_vt_kernel<<<8192, 256, 0, stream>>>(v_f32, vt_bf);

  // ---- stage 3: attention scores + softmax (batched over b,h) ----
  gemm(qt_bf, kt_bf, nullptr, S_t, nullptr, 256, 256, 64, 512, 512, 256, cB * cH, cH,
       (long long)256 * 512, 64, (long long)256 * 512, 64,
       (long long)8 * 65536, 65536, 0.125f, 0);
  softmax_rows<256><<<2048, 256, 0, stream>>>(S_t, at_bf, cB * cH * 256);
  gemm(qm_bf, km_bf, nullptr, S_m, nullptr, 64, 64, 64, 512, 512, 64, cB * cH, cH,
       (long long)64 * 512, 64, (long long)64 * 512, 64,
       (long long)8 * 4096, 4096, 0.125f, 0);
  softmax_rows<64><<<512, 256, 0, stream>>>(S_m, am_bf, cB * cH * 64);

  // ---- stage 4: apply time attention: tmp[b,h] = a_t[256x256] * vt^T ----
  gemm(at_bf, vt_bf, nullptr, tmp_f, nullptr, 256, 1024, 256, 256, 256, 1024,
       cB * cH, cB * cH, 0, 65536, 0, 262144, 0, 262144, 1.0f, 0);
  pack_tmp2_kernel<<<8192, 256, 0, stream>>>(tmp_f, tmp2_bf);

  // ---- stage 5: apply measure attention: attn2[b,h] = tmp2[4096x64] * a_m^T ----
  gemm(tmp2_bf, am_bf, nullptr, attn2_f, nullptr, 4096, 64, 64, 64, 64, 64,
       cB * cH, cB * cH, 0, 262144, 0, 4096, 0, 262144, 1.0f, 0);
  pack_attn_rows_kernel<<<8192, 256, 0, stream>>>(attn2_f, arow_bf);

  // ---- stage 6: output projection ----
  gemm(arow_bf, o_wT, o_b, attn_o, nullptr, cNTOK, 128, 128, 128, 128, 128,
       1, 1, 0,0,0,0,0,0, 1.0f, 0);

  // ---- stage 7: residual + LN1 ----
  add_layernorm<<<16384, 256, 0, stream>>>(x, attn_o, ln1_g, ln1_b, out1_f, out1_bf, cNTOK);

  // ---- stage 8: FFN (relu fused into GEMM1 epilogue, bf16 handoff) ----
  gemm(out1_bf, f1_wT, f1_b, nullptr, h1_bf, cNTOK, 512, 128, 128, 128, 512,
       1, 1, 0,0,0,0,0,0, 1.0f, 1);
  gemm(h1_bf, f2_wT, f2_b, ffn_f, nullptr, cNTOK, 128, 512, 512, 512, 128,
       1, 1, 0,0,0,0,0,0, 1.0f, 0);

  // ---- stage 9: residual + LN2 -> final output ----
  add_layernorm<<<16384, 256, 0, stream>>>(out1_f, ffn_f, ln2_g, ln2_b, outp, nullptr, cNTOK);
}
